// WindowAttentionBlock_36344013258881
// MI455X (gfx1250) — matrix operand order repaired and checked by reading, not compiled, not run on hardware
//
#include <hip/hip_runtime.h>
#include <hip/hip_bf16.h>

typedef __attribute__((ext_vector_type(16))) _Float16 v16h;
typedef __attribute__((ext_vector_type(8)))  float    v8f;

#define WMMA_F16(a, b, c) \
  __builtin_amdgcn_wmma_f32_16x16x32_f16(false, (a), false, (b), (short)0, (c), false, false)

// ---------------------------------------------------------------------------
// WMMA fragment loaders (wave32), this code's contiguous k order (A and B match).
// A (16x32 f16): lane m = lane&15; lanes 0-15 hold K 0..15,
//                lanes 16-31 hold K 16..31; element e -> K=kh+e.
// ---------------------------------------------------------------------------
__device__ inline v16h load_a_f16(const _Float16* __restrict__ base, int ld,
                                  int m0, int k0) {
  int lane = threadIdx.x & 31;
  const _Float16* p = base + (size_t)(m0 + (lane & 15)) * ld + k0 + ((lane >> 4) << 4);
  v16h a;
#pragma unroll
  for (int i = 0; i < 8; ++i) a[i] = p[i];
#pragma unroll
  for (int i = 0; i < 8; ++i) a[i + 8] = p[8 + i];
  return a;
}

// B (32x16 f16) from row-major K x N storage: lane n = lane&15;
// lanes 0-15 hold K=0..15, lanes 16-31 hold K=16..31 (element e -> K=kh+e).
__device__ inline v16h load_b_f16(const _Float16* __restrict__ base, int ld,
                                  int k0, int n0) {
  int lane = threadIdx.x & 31;
  const _Float16* p = base + (size_t)(k0 + ((lane >> 4) << 4)) * ld + n0 + (lane & 15);
  v16h b;
#pragma unroll
  for (int i = 0; i < 16; ++i) b[i] = p[(size_t)i * ld];
  return b;
}

// B[k][n] = W[n][k] with W row-major N x K: contiguous 32-byte load per lane.
__device__ inline v16h load_bT_f16(const _Float16* __restrict__ W, int ld,
                                   int n0, int k0) {
  int lane = threadIdx.x & 31;
  const _Float16* p = W + (size_t)(n0 + (lane & 15)) * ld + k0 + ((lane >> 4) << 4);
  v16h b;
#pragma unroll
  for (int i = 0; i < 16; ++i) b[i] = p[i];
  return b;
}

// ---------------------------------------------------------------------------
// Kernel 0: weight prep. f32 -> f16, fold q-scale (1/sqrt(32)) into the first
// 512 output features of qkv_w / qkv_b.
// ---------------------------------------------------------------------------
__global__ __launch_bounds__(256) void prep_weights(
    const float* __restrict__ qkv_w, const float* __restrict__ qkv_b,
    const float* __restrict__ proj_w,
    _Float16* __restrict__ qkvw16, float* __restrict__ qkvb_s,
    _Float16* __restrict__ projw16) {
  const float scale = 0.17677669529663687f;  // 32^-0.5
  int i = blockIdx.x * 256 + threadIdx.x;
  if (i < 786432) {                       // 1536*512
    float s = (i < 512 * 512) ? scale : 1.f;
    qkvw16[i] = (_Float16)(qkv_w[i] * s);
  } else if (i < 786432 + 262144) {       // 512*512
    int j = i - 786432;
    projw16[j] = (_Float16)proj_w[j];
  } else if (i < 786432 + 262144 + 1536) {
    int j = i - (786432 + 262144);
    qkvb_s[j] = qkv_b[j] * (j < 512 ? scale : 1.f);
  }
}

// ---------------------------------------------------------------------------
// Kernel 1: LayerNorm + cyclic shift + window partition -> xw f16
// xw layout: [B*64 windows][64 rows (49 valid, pad zeroed)][512]
// One wave32 per padded row; 16 channels per lane + shuffle reduction.
// ---------------------------------------------------------------------------
__global__ __launch_bounds__(256) void ln_partition(
    const float* __restrict__ x, const float* __restrict__ gamma,
    const float* __restrict__ beta, _Float16* __restrict__ xw) {
  int lane = threadIdx.x & 31;
  int row = blockIdx.x * 8 + (threadIdx.x >> 5);  // 0 .. 131071
  int n = row & 63;
  int w = (row >> 6) & 63;
  int b = row >> 12;
  _Float16* dst = xw + (size_t)row * 512;
  if (n >= 49) {
#pragma unroll
    for (int i = 0; i < 16; ++i) dst[lane * 16 + i] = (_Float16)0.f;
    return;
  }
  int ys = (w >> 3) * 7 + n / 7;
  int xs = (w & 7) * 7 + n % 7;
  int y = ys + 3; if (y >= 56) y -= 56;   // inverse of roll(-SHIFT)
  int xc = xs + 3; if (xc >= 56) xc -= 56;
  const float* src = x + ((size_t)b * 3136 + y * 56 + xc) * 512;
  float v[16], s = 0.f, s2 = 0.f;
#pragma unroll
  for (int i = 0; i < 16; ++i) {
    v[i] = src[lane * 16 + i];
    s += v[i];
    s2 += v[i] * v[i];
  }
#pragma unroll
  for (int m = 1; m < 32; m <<= 1) {
    s += __shfl_xor(s, m);
    s2 += __shfl_xor(s2, m);
  }
  float mu = s * (1.f / 512.f);
  float var = s2 * (1.f / 512.f) - mu * mu;
  float rstd = rsqrtf(var + 1e-5f);
#pragma unroll
  for (int i = 0; i < 16; ++i) {
    int c = lane * 16 + i;
    dst[c] = (_Float16)((v[i] - mu) * rstd * gamma[c] + beta[c]);
  }
}

// ---------------------------------------------------------------------------
// Kernel 2: QKV GEMM. Each wave: 16(M) x 64(N) f32 tile, K=512.
// Output scattered to qkv[window][part(q/k/v)][head][64][32] f16.
// q part pre-scaled via folded weights.
// ---------------------------------------------------------------------------
__global__ __launch_bounds__(256) void qkv_gemm(
    const _Float16* __restrict__ xw, const _Float16* __restrict__ Wt,
    const float* __restrict__ bias, _Float16* __restrict__ qkv) {
  int lane = threadIdx.x & 31;
  int wg = blockIdx.x * 8 + (threadIdx.x >> 5);  // 0 .. 196607
  int mt = wg / 24;
  int ng = wg - mt * 24;
  int col0 = ng * 64 + (lane & 15);
  v8f acc[4];
#pragma unroll
  for (int nt = 0; nt < 4; ++nt) {
    float bv = bias[col0 + nt * 16];
#pragma unroll
    for (int r = 0; r < 8; ++r) acc[nt][r] = bv;
  }
  const _Float16* A = xw + (size_t)mt * 16 * 512;
  for (int k0 = 0; k0 < 512; k0 += 32) {
    if (k0 + 32 < 512) __builtin_prefetch(A + k0 + 32, 0, 0);
    v16h a = load_a_f16(A, 512, 0, k0);
#pragma unroll
    for (int nt = 0; nt < 4; ++nt) {
      v16h b = load_bT_f16(Wt, 512, ng * 64 + nt * 16, k0);
      acc[nt] = WMMA_F16(a, b, acc[nt]);
    }
  }
  int rowg = mt * 16 + ((lane >> 4) << 3);  // global padded row of element r=0
  int w = rowg >> 6;                        // tiles never straddle windows
#pragma unroll
  for (int nt = 0; nt < 4; ++nt) {
    int col = col0 + nt * 16;
    int part = col >> 9;
    int cin = col & 511;
    int head = cin >> 5;
    int d = cin & 31;
    _Float16* basep = qkv + ((((size_t)w * 3 + part) * 16 + head) * 64) * 32 + d;
#pragma unroll
    for (int r = 0; r < 8; ++r) {
      int rw = (rowg + r) & 63;
      basep[(size_t)rw * 32] = (_Float16)acc[nt][r];
    }
  }
}

// ---------------------------------------------------------------------------
// Kernel 3: fused windowed attention. 1 block (4 waves) per (window, head).
// S = q@k^T (16 WMMA) -> +bias +mask -> softmax in registers -> P to LDS f16
// -> O = P@V (4 WMMA) -> /rowsum -> xo f16 [win][64][head*32+d].
// ---------------------------------------------------------------------------
__global__ __launch_bounds__(128) void win_attention(
    const _Float16* __restrict__ qkv, const float* __restrict__ bias_tab,
    _Float16* __restrict__ xo) {
  int blk = blockIdx.x;            // w*16 + head
  int w = blk >> 4;
  int head = blk & 15;
  int wimg = w & 63;
  int wh = wimg >> 3, ww = wimg & 7;
  int tid = threadIdx.x;
  int lane = tid & 31;
  int wv = tid >> 5;               // M-tile 0..3

  __shared__ float    s_bias[169];     // this head's rel-pos bias table
  __shared__ int      s_reg[49];       // region id per token (shift mask)
  __shared__ _Float16 s_P[64 * 72];    // softmax probs, padded stride

  for (int i = tid; i < 169; i += 128) s_bias[i] = bias_tab[i * 16 + head];
  for (int i = tid; i < 49; i += 128) {
    int ys = wh * 7 + i / 7;
    int xs = ww * 7 + i % 7;
    int ry = ys < 49 ? 0 : (ys < 53 ? 1 : 2);
    int rx = xs < 49 ? 0 : (xs < 53 ? 1 : 2);
    s_reg[i] = ry * 3 + rx;
  }
  __syncthreads();

  const _Float16* Q  = qkv + ((((size_t)w * 3 + 0) * 16 + head) * 64) * 32;
  const _Float16* Km = qkv + ((((size_t)w * 3 + 1) * 16 + head) * 64) * 32;
  const _Float16* V  = qkv + ((((size_t)w * 3 + 2) * 16 + head) * 64) * 32;

  // S = q @ k^T  (K-dim = head_dim = 32 -> one WMMA per 16x16 tile)
  v16h qa = load_a_f16(Q, 32, wv * 16, 0);
  v8f s[4];
#pragma unroll
  for (int nt = 0; nt < 4; ++nt) {
    v16h kb = load_bT_f16(Km, 32, nt * 16, 0);
    v8f z = {};
    s[nt] = WMMA_F16(qa, kb, z);
  }

  int rowbase = wv * 16 + ((lane >> 4) << 3);  // row of element r=0 (C layout)
  int col0 = lane & 15;

  // bias + shift-mask, invalid columns -> -1e30
#pragma unroll
  for (int nt = 0; nt < 4; ++nt) {
    int col = nt * 16 + col0;
#pragma unroll
    for (int r = 0; r < 8; ++r) {
      int row = rowbase + r;
      float val = -1e30f;
      if (col < 49 && row < 49) {
        int dy = row / 7 - col / 7 + 6;
        int dx = row % 7 - col % 7 + 6;
        float msk = (s_reg[row] != s_reg[col]) ? -100.f : 0.f;
        val = s[nt][r] + s_bias[dy * 13 + dx] + msk;
      }
      s[nt][r] = val;
    }
  }

  // row softmax: C-layout rows stripe across the 16-lane half-groups, so
  // xor-shuffles with masks 1,2,4,8 reduce exactly one row.
  float rsum[8];
#pragma unroll
  for (int r = 0; r < 8; ++r) {
    float m = fmaxf(fmaxf(s[0][r], s[1][r]), fmaxf(s[2][r], s[3][r]));
#pragma unroll
    for (int msk = 1; msk < 16; msk <<= 1) m = fmaxf(m, __shfl_xor(m, msk));
    float sum = 0.f;
#pragma unroll
    for (int nt = 0; nt < 4; ++nt) {
      float p = __expf(s[nt][r] - m);
      s[nt][r] = p;
      sum += p;
    }
#pragma unroll
    for (int msk = 1; msk < 16; msk <<= 1) sum += __shfl_xor(sum, msk);
    rsum[r] = sum;
  }

  // P -> LDS (f16) in plain row-major so it can be re-read as A fragments
#pragma unroll
  for (int nt = 0; nt < 4; ++nt)
#pragma unroll
    for (int r = 0; r < 8; ++r)
      s_P[(rowbase + r) * 72 + nt * 16 + col0] = (_Float16)s[nt][r];
  __syncthreads();

  // O = P @ V   (M=16, N=32, K=64 -> 2x2 WMMA)
#pragma unroll
  for (int dt = 0; dt < 2; ++dt) {
    v8f o = {};
#pragma unroll
    for (int kk = 0; kk < 2; ++kk) {
      v16h pa = load_a_f16(s_P, 72, wv * 16, kk * 32);
      v16h vb = load_b_f16(V, 32, kk * 32, dt * 16);
      o = WMMA_F16(pa, vb, o);
    }
    _Float16* dst = xo + ((size_t)w * 64 + rowbase) * 512 + head * 32 + dt * 16 + col0;
#pragma unroll
    for (int r = 0; r < 8; ++r)
      dst[(size_t)r * 512] = (_Float16)(o[r] / rsum[r]);
  }
}

// ---------------------------------------------------------------------------
// Kernel 4: out-projection GEMM + window-reverse + reverse shift -> d_out f32
// ---------------------------------------------------------------------------
__global__ __launch_bounds__(256) void proj_gemm(
    const _Float16* __restrict__ xo, const _Float16* __restrict__ Wt,
    const float* __restrict__ bias, float* __restrict__ out) {
  int lane = threadIdx.x & 31;
  int wg = blockIdx.x * 8 + (threadIdx.x >> 5);  // 0 .. 65535
  int mt = wg >> 3;
  int ng = wg & 7;
  int col0 = ng * 64 + (lane & 15);
  v8f acc[4];
#pragma unroll
  for (int nt = 0; nt < 4; ++nt) {
    float bv = bias[col0 + nt * 16];
#pragma unroll
    for (int r = 0; r < 8; ++r) acc[nt][r] = bv;
  }
  const _Float16* A = xo + (size_t)mt * 16 * 512;
  for (int k0 = 0; k0 < 512; k0 += 32) {
    if (k0 + 32 < 512) __builtin_prefetch(A + k0 + 32, 0, 0);
    v16h a = load_a_f16(A, 512, 0, k0);
#pragma unroll
    for (int nt = 0; nt < 4; ++nt) {
      v16h b = load_bT_f16(Wt, 512, ng * 64 + nt * 16, k0);
      acc[nt] = WMMA_F16(a, b, acc[nt]);
    }
  }
  int rowg = mt * 16 + ((lane >> 4) << 3);
  int w = rowg >> 6;
  int b = w >> 6;
  int wi = w & 63;
  int wh = wi >> 3, ww = wi & 7;
#pragma unroll
  for (int r = 0; r < 8; ++r) {
    int n = (rowg + r) & 63;
    if (n >= 49) continue;                 // padded rows are dropped
    int ys = wh * 7 + n / 7;
    int xs = ww * 7 + n % 7;
    int y = ys + 3; if (y >= 56) y -= 56;  // roll(+SHIFT)
    int xc = xs + 3; if (xc >= 56) xc -= 56;
    float* drow = out + ((size_t)b * 3136 + y * 56 + xc) * 512;
#pragma unroll
    for (int nt = 0; nt < 4; ++nt) drow[col0 + nt * 16] = acc[nt][r];
  }
}

// ---------------------------------------------------------------------------
extern "C" void kernel_launch(void* const* d_in, const int* in_sizes, int n_in,
                              void* d_out, int out_size, void* d_ws, size_t ws_size,
                              hipStream_t stream) {
  const float* x        = (const float*)d_in[0];
  const float* gamma    = (const float*)d_in[1];
  const float* beta     = (const float*)d_in[2];
  const float* qkv_w    = (const float*)d_in[3];
  const float* qkv_b    = (const float*)d_in[4];
  const float* proj_w   = (const float*)d_in[5];
  const float* proj_b   = (const float*)d_in[6];
  const float* bias_tab = (const float*)d_in[7];
  float* out = (float*)d_out;

  char* ws = (char*)d_ws;
  size_t off = 0;
  auto take = [&](size_t bytes) {
    char* p = ws + off;
    off += (bytes + 255) & ~(size_t)255;
    return p;
  };
  _Float16* qkvw16 = (_Float16*)take(786432ull * 2);                 // 1.5 MB
  _Float16* projw16 = (_Float16*)take(262144ull * 2);                // 0.5 MB
  float*    qkvb_s  = (float*)take(1536ull * 4);
  _Float16* xw  = (_Float16*)take(131072ull * 512 * 2);              // 134 MB
  _Float16* qkv = (_Float16*)take(2048ull * 3 * 16 * 64 * 32 * 2);   // 403 MB
  _Float16* xo  = (_Float16*)take(131072ull * 512 * 2);              // 134 MB
  (void)ws_size; (void)in_sizes; (void)n_in; (void)out_size;

  prep_weights<<<(786432 + 262144 + 1536 + 255) / 256, 256, 0, stream>>>(
      qkv_w, qkv_b, proj_w, qkvw16, qkvb_s, projw16);

  ln_partition<<<131072 / 8, 256, 0, stream>>>(x, gamma, beta, xw);

  // M-tiles = 131072/16 = 8192, N-groups = 1536/64 = 24 -> 196608 waves
  qkv_gemm<<<196608 / 8, 256, 0, stream>>>(xw, qkvw16, qkvb_s, qkv);

  win_attention<<<2048 * 16, 128, 0, stream>>>(qkv, bias_tab, xo);

  // M-tiles = 8192, N-groups = 512/64 = 8 -> 65536 waves
  proj_gemm<<<65536 / 8, 256, 0, stream>>>(xo, projw16, proj_b, out);
}
